// MultiLatentAttention_2138893714152
// MI455X (gfx1250) — matrix operand order repaired and checked by reading, not compiled, not run on hardware
//
#include <hip/hip_runtime.h>
#include <hip/hip_bf16.h>

// ---------------- problem constants ----------------
constexpr int B_ = 2, T_ = 2048, E_ = 2048, H_ = 16, D_ = 128, LOW_ = 512;
constexpr int DQK = 2 * D_;   // 256 (q/k head dim after concat)
constexpr int M_ = B_ * T_;   // 4096 rows in all projection GEMMs

typedef __bf16 bf16_t;
typedef __attribute__((ext_vector_type(4)))  __bf16 v4bf;
typedef __attribute__((ext_vector_type(8)))  __bf16 v8bf;
typedef __attribute__((ext_vector_type(16))) __bf16 v16bf;
typedef __attribute__((ext_vector_type(8)))  float  v8f;
typedef int i32x4 __attribute__((vector_size(16)));  // matches builtin param

union BF16Frag { v16bf v; v8bf h[2]; };

#define GAS __attribute__((address_space(1)))
#define LAS __attribute__((address_space(3)))

// ---- CDNA5 async global->LDS copy (ASYNCcnt-tracked), with fallback ----
#if __has_builtin(__builtin_amdgcn_global_load_async_to_lds_b128)
#define HAVE_ASYNC_LDS 1
#endif

__device__ __forceinline__ void cp16_async(const void* g, void* l) {
#if defined(HAVE_ASYNC_LDS)
  __builtin_amdgcn_global_load_async_to_lds_b128(
      (GAS i32x4*)g, (LAS i32x4*)l, 0, 0);
#else
  *(v8bf*)l = *(const v8bf*)g;
#endif
}

__device__ __forceinline__ void wait_async() {
#if __has_builtin(__builtin_amdgcn_s_wait_asynccnt)
  __builtin_amdgcn_s_wait_asynccnt(0);
#elif defined(HAVE_ASYNC_LDS)
  asm volatile("s_wait_asynccnt 0x0" ::: "memory");
#endif
}

__device__ __forceinline__ v8f wmma_bf16(v16bf a, v16bf b, v8f c) {
  return __builtin_amdgcn_wmma_f32_16x16x32_bf16(false, a, false, b,
                                                 (short)0, c, false, false);
}

// =====================================================================
// Conversion kernels (run once per call; all GEMM traffic becomes bf16)
// =====================================================================
__global__ void cvt_f32_bf16(const float* __restrict__ in,
                             bf16_t* __restrict__ out) {
  size_t i = ((size_t)blockIdx.x * blockDim.x + threadIdx.x) * 4;
  float4 f = *(const float4*)(in + i);
  v4bf hv = { (__bf16)f.x, (__bf16)f.y, (__bf16)f.z, (__bf16)f.w };
  *(v4bf*)(out + i) = hv;
}

// W fp32 [K][N]  ->  bf16 [N][K]  (coalesced via 32x32 LDS tile)
__global__ __launch_bounds__(256) void transpose_cvt(
    const float* __restrict__ in, bf16_t* __restrict__ out, int K, int N) {
  __shared__ float tile[32][33];
  int n0 = blockIdx.x * 32, k0 = blockIdx.y * 32;
  int tx = threadIdx.x & 31, ty = threadIdx.x >> 5;  // ty: 0..7
#pragma unroll
  for (int i = 0; i < 4; ++i)
    tile[ty + i * 8][tx] = in[(size_t)(k0 + ty + i * 8) * N + n0 + tx];
  __syncthreads();
#pragma unroll
  for (int i = 0; i < 4; ++i)
    out[(size_t)(n0 + ty + i * 8) * K + k0 + tx] = (__bf16)tile[tx][ty + i * 8];
}

// =====================================================================
// GEMM: C[M,N] = A[M,K] @ Bt[N,K]^T, bf16 operands, f32 accumulate.
// 128x128x32 block tile, double-buffered LDS filled by async copies.
// 8 wave32 waves in 4(M) x 2(N) grid; each wave 32x64 = 8 accumulators.
// =====================================================================
constexpr int BM = 128, BN = 128, BK = 32;
constexpr int LDT = BK + 8;  // 40 halves: 80B rows, 16B aligned

template <typename OutT>
__global__ __launch_bounds__(256) void gemm_bf16_tn(
    const bf16_t* __restrict__ A, const bf16_t* __restrict__ Bt,
    OutT* __restrict__ C, int M, int N, int K) {
  __shared__ bf16_t As[2][BM][LDT];
  __shared__ bf16_t Bs[2][BN][LDT];

  const int tid  = threadIdx.x;
  const int lane = tid & 31;
  const int wid  = tid >> 5;
  const int wm   = wid & 3;          // 32-row band
  const int wn   = wid >> 2;         // 64-col band
  const int m0   = blockIdx.y * BM;
  const int n0   = blockIdx.x * BN;
  const int half = (lane < 16) ? 0 : 1;
  const int l16  = lane & 15;

  v8f acc[2][4];
  for (int i = 0; i < 2; ++i)
    for (int j = 0; j < 4; ++j)
      for (int e = 0; e < 8; ++e) acc[i][j][e] = 0.0f;

  // each thread copies 2 x 16B chunks per operand per tile
  const int c0 = tid, c1 = tid + 256;           // chunk ids 0..511
  const int r0 = c0 >> 2, p0 = (c0 & 3) << 3;   // row, half-offset
  const int r1 = c1 >> 2, p1 = (c1 & 3) << 3;

  auto load_tile = [&](int buf, int k0) {
    cp16_async(A  + (size_t)(m0 + r0) * K + k0 + p0, &As[buf][r0][p0]);
    cp16_async(A  + (size_t)(m0 + r1) * K + k0 + p1, &As[buf][r1][p1]);
    cp16_async(Bt + (size_t)(n0 + r0) * K + k0 + p0, &Bs[buf][r0][p0]);
    cp16_async(Bt + (size_t)(n0 + r1) * K + k0 + p1, &Bs[buf][r1][p1]);
  };

  const int nk = K / BK;
  load_tile(0, 0);

  for (int kt = 0; kt < nk; ++kt) {
    wait_async();
    __syncthreads();
    if (kt + 1 < nk) load_tile((kt + 1) & 1, (kt + 1) * BK);

    const int buf = kt & 1;
    BF16Frag afrag[2], bfrag[4];
#pragma unroll
    for (int mt = 0; mt < 2; ++mt) {
      int m  = wm * 32 + mt * 16 + l16;
      int kb = half * 8;   // A lanes 0-15: K0..7/16..23; 16-31: K8..15/24..31
      afrag[mt].h[0] = *(const v8bf*)&As[buf][m][kb];
      afrag[mt].h[1] = *(const v8bf*)&As[buf][m][kb + 16];
    }
#pragma unroll
    for (int nt = 0; nt < 4; ++nt) {
      int n  = wn * 64 + nt * 16 + l16;
      int kb = half * 8;   // B lanes 0-15: K0..7/16..23; 16-31: K8..15/24..31
      bfrag[nt].h[0] = *(const v8bf*)&Bs[buf][n][kb];
      bfrag[nt].h[1] = *(const v8bf*)&Bs[buf][n][kb + 16];
    }
#pragma unroll
    for (int mt = 0; mt < 2; ++mt)
#pragma unroll
      for (int nt = 0; nt < 4; ++nt)
        acc[mt][nt] = wmma_bf16(afrag[mt].v, bfrag[nt].v, acc[mt][nt]);
  }

#pragma unroll
  for (int mt = 0; mt < 2; ++mt)
#pragma unroll
    for (int nt = 0; nt < 4; ++nt)
#pragma unroll
      for (int j = 0; j < 8; ++j) {
        int row = m0 + wm * 32 + mt * 16 + half * 8 + j;
        int col = n0 + wn * 64 + nt * 16 + l16;
        C[(size_t)row * N + col] = (OutT)acc[mt][nt][j];
      }
}

// =====================================================================
// Epilogues: rope + concat into bf16 q/k (B,H,T,256); v transpose
// =====================================================================
__global__ void build_qk_rope(const bf16_t* __restrict__ Cpart,
                              const bf16_t* __restrict__ Rpart,
                              bf16_t* __restrict__ out) {
  size_t tid = (size_t)blockIdx.x * blockDim.x + threadIdx.x; // B*H*T*64 threads
  int    i   = (int)(tid & 63);
  size_t r   = tid >> 6;
  int    t   = (int)(r % T_);
  int    bh  = (int)(r / T_);
  int    b   = bh / H_, h = bh % H_;
  size_t src = ((size_t)b * T_ + t) * E_ + (size_t)h * D_;   // (b,t,h,d)
  size_t dst = ((size_t)bh * T_ + t) * DQK;                  // (b,h,t,:)
  out[dst + 2 * i]     = Cpart[src + 2 * i];
  out[dst + 2 * i + 1] = Cpart[src + 2 * i + 1];
  float x1 = (float)Rpart[src + 2 * i];
  float x2 = (float)Rpart[src + 2 * i + 1];
  float freq = __powf(10000.0f, -(float)i * (1.0f / 64.0f));
  float th = (float)t * freq, sn, cs;
  __sincosf(th, &sn, &cs);
  out[dst + D_ + i]      = (__bf16)(x1 * cs - x2 * sn);
  out[dst + D_ + 64 + i] = (__bf16)(x1 * sn + x2 * cs);
}

__global__ void build_vT(const bf16_t* __restrict__ Vtmp, bf16_t* __restrict__ vT) {
  size_t tid = (size_t)blockIdx.x * blockDim.x + threadIdx.x; // B*H*D*T threads
  int    t   = (int)(tid % T_);
  size_t r   = tid / T_;
  int    d   = (int)(r % D_);
  int    bh  = (int)(r / D_);
  int    b   = bh / H_, h = bh % H_;
  vT[tid] = Vtmp[((size_t)b * T_ + t) * E_ + (size_t)h * D_ + d];
}

// =====================================================================
// Flash attention with causal mask. 64 q-rows per block for one (b,h).
// Q fragments in registers; K/V staged via async copies; WMMA for
// QK^T and PV; online softmax through LDS. Output bf16 (B,T,E).
// =====================================================================
constexpr int QT = 64, KT = 64;
constexpr int LDS_K = DQK + 8;  // 264 halves
constexpr int LDS_V = KT + 8;   // 72 halves
constexpr int LDS_P = KT + 8;   // 72 halves

__global__ __launch_bounds__(256) void flash_attn(
    const bf16_t* __restrict__ q, const bf16_t* __restrict__ k,
    const bf16_t* __restrict__ vT, bf16_t* __restrict__ o) {
  __shared__ bf16_t Ks[KT][LDS_K];
  __shared__ bf16_t Vs[D_][LDS_V];   // V transposed: [dim][key]
  __shared__ bf16_t Ps[QT][LDS_P];   // scores, then probabilities
  __shared__ float row_max[QT], row_sum[QT], row_corr[QT];

  const int tid = threadIdx.x, lane = tid & 31, wid = tid >> 5;
  const int half = (lane < 16) ? 0 : 1, l16 = lane & 15;
  const int qt = blockIdx.x;
  const int bh = blockIdx.y;
  const int b  = bh / H_, h = bh % H_;
  const size_t qoff   = ((size_t)bh * T_ + (size_t)qt * QT) * DQK;
  const bf16_t* kbase = k + (size_t)bh * T_ * DQK;
  const bf16_t* vbase = vT + (size_t)bh * D_ * T_;
  const float scale = 0.0625f;  // 1/sqrt(2*D)

  const int swm = wid & 3,  swn = wid >> 2;  // S-phase: 4x2, 16-row x 32-key
  const int owm = wid >> 2, own = wid & 3;   // O-phase: 2x4, 32-row x 32-dim

  BF16Frag qf[8];
  {
    const bf16_t* qrow = q + qoff + (size_t)(swm * 16 + l16) * DQK;
#pragma unroll
    for (int kk = 0; kk < 8; ++kk) {
      int kb = kk * 32 + half * 8;
      qf[kk].h[0] = *(const v8bf*)(qrow + kb);
      qf[kk].h[1] = *(const v8bf*)(qrow + kb + 16);
    }
  }

  v8f oacc[2][2];
  for (int i = 0; i < 2; ++i)
    for (int j = 0; j < 2; ++j)
      for (int e = 0; e < 8; ++e) oacc[i][j][e] = 0.0f;
  if (tid < QT) { row_max[tid] = -1e30f; row_sum[tid] = 0.0f; }
  __syncthreads();

  for (int kt = 0; kt <= qt; ++kt) {
    // ---- stage K tile (64x256) and V tile (128x64) via async copies ----
#pragma unroll
    for (int p = 0; p < 8; ++p) {
      int c = tid + p * 256;
      int r = c >> 5, pos = (c & 31) << 3;
      cp16_async(kbase + (size_t)(kt * KT + r) * DQK + pos, &Ks[r][pos]);
    }
#pragma unroll
    for (int p = 0; p < 4; ++p) {
      int c = tid + p * 256;
      int r = c >> 3, pos = (c & 7) << 3;
      cp16_async(vbase + (size_t)r * T_ + kt * KT + pos, &Vs[r][pos]);
    }
    wait_async();
    __syncthreads();

    // ---- S = Q @ K^T ----
    v8f sacc[2];
    for (int j = 0; j < 2; ++j)
      for (int e = 0; e < 8; ++e) sacc[j][e] = 0.0f;
#pragma unroll
    for (int kk = 0; kk < 8; ++kk) {
#pragma unroll
      for (int nt = 0; nt < 2; ++nt) {
        BF16Frag bf;
        int n  = swn * 32 + nt * 16 + l16;
        int kb = kk * 32 + half * 8;
        bf.h[0] = *(const v8bf*)&Ks[n][kb];
        bf.h[1] = *(const v8bf*)&Ks[n][kb + 16];
        sacc[nt] = wmma_bf16(qf[kk].v, bf.v, sacc[nt]);
      }
    }
#pragma unroll
    for (int nt = 0; nt < 2; ++nt)
#pragma unroll
      for (int j = 0; j < 8; ++j) {
        int r  = swm * 16 + half * 8 + j;
        int cl = swn * 32 + nt * 16 + l16;
        float s = sacc[nt][j] * scale;
        if (kt * KT + cl > qt * QT + r) s = -1e30f;
        Ps[r][cl] = (__bf16)s;
      }
    __syncthreads();

    // ---- online softmax: one thread per q row ----
    if (tid < QT) {
      float mo = row_max[tid], mx = mo;
      for (int c2 = 0; c2 < KT; ++c2) mx = fmaxf(mx, (float)Ps[tid][c2]);
      float corr = __expf(mo - mx);
      float sum  = row_sum[tid] * corr;
      for (int c2 = 0; c2 < KT; ++c2) {
        float p = __expf((float)Ps[tid][c2] - mx);
        sum += p;
        Ps[tid][c2] = (__bf16)p;
      }
      row_max[tid] = mx; row_sum[tid] = sum; row_corr[tid] = corr;
    }
    __syncthreads();

    // ---- rescale O, then O += P @ V ----
#pragma unroll
    for (int mt = 0; mt < 2; ++mt)
#pragma unroll
      for (int nt = 0; nt < 2; ++nt)
#pragma unroll
        for (int j = 0; j < 8; ++j) {
          int r = owm * 32 + mt * 16 + half * 8 + j;
          oacc[mt][nt][j] *= row_corr[r];
        }
#pragma unroll
    for (int ks = 0; ks < 2; ++ks) {
      BF16Frag pf[2], vf[2];
#pragma unroll
      for (int mt = 0; mt < 2; ++mt) {
        int m  = owm * 32 + mt * 16 + l16;
        int kb = ks * 32 + half * 8;
        pf[mt].h[0] = *(const v8bf*)&Ps[m][kb];
        pf[mt].h[1] = *(const v8bf*)&Ps[m][kb + 16];
      }
#pragma unroll
      for (int nt = 0; nt < 2; ++nt) {
        int n  = own * 32 + nt * 16 + l16;
        int kb = ks * 32 + half * 8;
        vf[nt].h[0] = *(const v8bf*)&Vs[n][kb];
        vf[nt].h[1] = *(const v8bf*)&Vs[n][kb + 16];
      }
#pragma unroll
      for (int mt = 0; mt < 2; ++mt)
#pragma unroll
        for (int nt = 0; nt < 2; ++nt)
          oacc[mt][nt] = wmma_bf16(pf[mt].v, vf[nt].v, oacc[mt][nt]);
    }
    __syncthreads();
  }

  // ---- normalize & store bf16 to (B,T,E) = (b, t, h*D + d) ----
#pragma unroll
  for (int mt = 0; mt < 2; ++mt)
#pragma unroll
    for (int nt = 0; nt < 2; ++nt)
#pragma unroll
      for (int j = 0; j < 8; ++j) {
        int r    = owm * 32 + mt * 16 + half * 8 + j;
        int dcol = own * 32 + nt * 16 + l16;
        float val = oacc[mt][nt][j] / row_sum[r];
        size_t trow = (size_t)b * T_ + (size_t)qt * QT + r;
        o[trow * E_ + (size_t)h * D_ + dcol] = (__bf16)val;
      }
}

// =====================================================================
// Launch: convert once to bf16 (+weight transpose), 8 WMMA GEMMs,
// epilogues, flash attention, final fp32 GEMM. ws ~200 MB.
// =====================================================================
extern "C" void kernel_launch(void* const* d_in, const int* in_sizes, int n_in,
                              void* d_out, int out_size, void* d_ws, size_t ws_size,
                              hipStream_t stream) {
  const float* x    = (const float*)d_in[0];
  const float* Wdq  = (const float*)d_in[1];
  const float* Wuq  = (const float*)d_in[2];
  const float* Wqr  = (const float*)d_in[3];
  const float* Wdkv = (const float*)d_in[4];
  const float* Wuk  = (const float*)d_in[5];
  const float* Wkr  = (const float*)d_in[6];
  const float* Wuv  = (const float*)d_in[7];
  const float* Wo   = (const float*)d_in[8];
  float* out = (float*)d_out;

  const size_t ME = (size_t)M_ * E_;  // 8,388,608
  const size_t EE = (size_t)E_ * E_;  // 4,194,304
  const size_t EL = (size_t)E_ * LOW_;

  bf16_t* p = (bf16_t*)d_ws;
  bf16_t* xb     = p; p += ME;
  bf16_t* wdq_t  = p; p += EE;
  bf16_t* wuq_t  = p; p += EE;
  bf16_t* wqr_t  = p; p += EE;
  bf16_t* wkr_t  = p; p += EE;
  bf16_t* wo_t   = p; p += EE;
  bf16_t* wdkv_t = p; p += EL;
  bf16_t* wuk_t  = p; p += EL;
  bf16_t* wuv_t  = p; p += EL;
  bf16_t* buf0   = p; p += ME;  // Cq / Ckv / attn_out
  bf16_t* buf1   = p; p += ME;  // Qc / Kc / V (pre-epilogue)
  bf16_t* buf2   = p; p += ME;  // Qr / Kr
  bf16_t* qb     = p; p += (size_t)B_ * H_ * T_ * DQK;
  bf16_t* kb     = p; p += (size_t)B_ * H_ * T_ * DQK;
  bf16_t* vb     = p; p += (size_t)B_ * H_ * D_ * T_;

  dim3 blk(256);
  dim3 gFull(E_ / BN, M_ / BM);    // (16, 32)
  dim3 gLow(LOW_ / BN, M_ / BM);   // (4, 32)
  const unsigned nQK = (unsigned)(((size_t)B_ * H_ * T_ * 64) / 256);
  const unsigned nV  = (unsigned)(((size_t)B_ * H_ * D_ * T_) / 256);

  // ---- one-time bf16 conversion / weight transposition ----
  cvt_f32_bf16<<<(unsigned)(ME / 1024), blk, 0, stream>>>(x, xb);
  transpose_cvt<<<dim3(E_/32,  E_/32),  blk, 0, stream>>>(Wdq,  wdq_t,  E_,  E_);
  transpose_cvt<<<dim3(E_/32,  E_/32),  blk, 0, stream>>>(Wuq,  wuq_t,  E_,  E_);
  transpose_cvt<<<dim3(E_/32,  E_/32),  blk, 0, stream>>>(Wqr,  wqr_t,  E_,  E_);
  transpose_cvt<<<dim3(E_/32,  E_/32),  blk, 0, stream>>>(Wkr,  wkr_t,  E_,  E_);
  transpose_cvt<<<dim3(E_/32,  E_/32),  blk, 0, stream>>>(Wo,   wo_t,   E_,  E_);
  transpose_cvt<<<dim3(LOW_/32, E_/32), blk, 0, stream>>>(Wdkv, wdkv_t, E_,  LOW_);
  transpose_cvt<<<dim3(E_/32, LOW_/32), blk, 0, stream>>>(Wuk,  wuk_t,  LOW_, E_);
  transpose_cvt<<<dim3(E_/32, LOW_/32), blk, 0, stream>>>(Wuv,  wuv_t,  LOW_, E_);

  // ---- projections (all bf16 WMMA, async double-buffered) ----
  gemm_bf16_tn<bf16_t><<<gFull, blk, 0, stream>>>(xb,   wdq_t, buf0, M_, E_, E_);
  gemm_bf16_tn<bf16_t><<<gFull, blk, 0, stream>>>(buf0, wuq_t, buf1, M_, E_, E_);
  gemm_bf16_tn<bf16_t><<<gFull, blk, 0, stream>>>(buf0, wqr_t, buf2, M_, E_, E_);
  build_qk_rope<<<nQK, blk, 0, stream>>>(buf1, buf2, qb);
  gemm_bf16_tn<bf16_t><<<gLow,  blk, 0, stream>>>(xb,   wdkv_t, buf0, M_, LOW_, E_);
  gemm_bf16_tn<bf16_t><<<gFull, blk, 0, stream>>>(buf0, wuk_t,  buf1, M_, E_, LOW_);
  gemm_bf16_tn<bf16_t><<<gFull, blk, 0, stream>>>(xb,   wkr_t,  buf2, M_, E_, E_);
  build_qk_rope<<<nQK, blk, 0, stream>>>(buf1, buf2, kb);
  gemm_bf16_tn<bf16_t><<<gFull, blk, 0, stream>>>(buf0, wuv_t,  buf1, M_, E_, LOW_);
  build_vT<<<nV, blk, 0, stream>>>(buf1, vb);

  // ---- attention -> bf16 (B,T,E), then out = attn @ Wo (fp32) ----
  flash_attn<<<dim3(T_ / QT, B_ * H_), blk, 0, stream>>>(qb, kb, vb, buf0);
  gemm_bf16_tn<float><<<gFull, blk, 0, stream>>>(buf0, wo_t, out, M_, E_, E_);
}
